// BusterModel_46651934769682
// MI455X (gfx1250) — matrix operand order repaired and checked by reading, not compiled, not run on hardware
//
#include <hip/hip_runtime.h>
#include <hip/hip_bf16.h>
#include <math.h>

typedef __attribute__((ext_vector_type(16))) _Float16 v16h;
typedef __attribute__((ext_vector_type(8)))  float    v8f;
typedef unsigned int v4ui __attribute__((ext_vector_type(4)));
typedef int          v8i  __attribute__((ext_vector_type(8)));
typedef int          v4i  __attribute__((ext_vector_type(4)));

#define HW   1600      // 40*40 spatial positions
#define CC   896       // channels
#define BB   8         // batch
#define NROW (BB*HW)   // 12800 lines per orientation

// output element offsets (f32)
#define O_X2CAT  22937600u   // 8*1792*1600
#define O_VAL1   45875200u
#define O_VAL2   46003200u

#if __has_builtin(__builtin_amdgcn_tensor_load_to_lds) && __has_builtin(__builtin_amdgcn_s_wait_tensorcnt)
#define USE_TDM 1
#else
#define USE_TDM 0
#endif

// ---------------------------------------------------------------------------
// Kernel 1: batched GEMM  xco[b] = A(1600x896) * B(896x1600) with WMMA f16->f32
//   A[i,c] = x1[b,c,i]  (i contiguous in memory)
//   B[c,j] = x2[b,c,j]  (j contiguous in memory)
// Block tile 64x64, 4 waves, each wave 32x32 (2x2 WMMA 16x16x32 tiles).
// Tile staging via Tensor Data Mover (tensor_load_to_lds + s_wait_tensorcnt)
// when available; direct global->LDS staging otherwise.
// ---------------------------------------------------------------------------
__global__ __launch_bounds__(128)
void corr_gemm_wmma(const float* __restrict__ x1, const float* __restrict__ x2,
                    float* __restrict__ M)
{
    __shared__ __align__(16) _Float16 As[64 * 32];  // As[i_loc*32 + c_loc]
    __shared__ __align__(16) _Float16 Bs[64 * 32];  // Bs[j_loc*32 + c_loc]
#if USE_TDM
    __shared__ __align__(16) float stgA[32 * 64];   // stgA[c_loc*64 + i_loc] (TDM row order)
    __shared__ __align__(16) float stgB[32 * 64];
#endif

    const int b   = blockIdx.z;
    const int m0  = blockIdx.y * 64;
    const int n0  = blockIdx.x * 64;
    const int t   = threadIdx.x;        // 0..127
    const int wave = t >> 5;            // 0..3
    const int lane = t & 31;
    const int l16  = lane & 15;
    const int hi   = lane >> 4;         // lane group 0/1
    const int wm   = (wave >> 1) * 32;  // wave row offset in block tile
    const int wn   = (wave & 1) * 32;   // wave col offset

    const float* Ab = x1 + (size_t)b * CC * HW;
    const float* Bg = x2 + (size_t)b * CC * HW;

    v8f acc[2][2];
    const v8f vzero = {0.f,0.f,0.f,0.f,0.f,0.f,0.f,0.f};
#pragma unroll
    for (int mi = 0; mi < 2; ++mi)
#pragma unroll
        for (int ni = 0; ni < 2; ++ni) acc[mi][ni] = vzero;

#if USE_TDM
    // LDS byte offsets: low 32 bits of the generic (flat) LDS address are the
    // LDS offset per the CDNA5 aperture mapping.
    const unsigned ldsA = (unsigned)(size_t)(void*)stgA;
    const unsigned ldsB = (unsigned)(size_t)(void*)stgB;
    // D# group 1 (shared by A and B tiles): data_size=4B, tensor 1600x896,
    // tile 64x32, dim0 stride 1600 elements. workgroup_mask=0 (not in cluster).
    const v8i g1 = { (int)(2u << 16),     // [17:16] data_size = 2 (4 bytes)
                     (int)(1600u << 16),  // tensor_dim0[15:0] in bits 63:48
                     (int)(896u << 16),   // tensor_dim0 hi=0 | tensor_dim1 lo16
                     (int)(64u << 16),    // tensor_dim1 hi=0 | tile_dim0 = 64
                     (int)32,             // tile_dim1 = 32, tile_dim2 = 0
                     (int)1600,           // tensor_dim0_stride[31:0]
                     0, 0 };              // stride0 hi / stride1 (unused, 2D)
    const v4i gz4 = {0, 0, 0, 0};
    const v8i gz8 = {0, 0, 0, 0, 0, 0, 0, 0};
#endif

    for (int k0 = 0; k0 < CC; k0 += 32) {
#if USE_TDM
        if (wave == 0) {
            unsigned long long ga = (unsigned long long)(const void*)(Ab + (size_t)k0 * HW + m0);
            unsigned long long gb = (unsigned long long)(const void*)(Bg + (size_t)k0 * HW + n0);
            v4ui g0a = { 1u, ldsA, (unsigned)ga,
                         (unsigned)((ga >> 32) & 0x1ffffffu) | (2u << 30) };
            v4ui g0b = { 1u, ldsB, (unsigned)gb,
                         (unsigned)((gb >> 32) & 0x1ffffffu) | (2u << 30) };
            __builtin_amdgcn_tensor_load_to_lds(g0a, g1, gz4, gz4, gz8, 0);
            __builtin_amdgcn_tensor_load_to_lds(g0b, g1, gz4, gz4, gz8, 0);
            __builtin_amdgcn_s_wait_tensorcnt(0);
        }
        __syncthreads();
        // convert staged f32 tiles -> f16 fragment buffers ([row][k], k contiguous)
#pragma unroll
        for (int q = 0; q < 16; ++q) {
            int e    = q * 128 + t;
            int col  = e & 63;    // i_loc / j_loc
            int crow = e >> 6;    // c_loc
            As[col * 32 + crow] = (_Float16)stgA[crow * 64 + col];
            Bs[col * 32 + crow] = (_Float16)stgB[crow * 64 + col];
        }
#else
        // stage 64x32 tiles (2048 f32 each), convert f32 -> f16 into LDS
#pragma unroll
        for (int q = 0; q < 16; ++q) {
            int e    = q * 128 + t;
            int col  = e & 63;    // i_loc / j_loc (contiguous in global)
            int crow = e >> 6;    // c_loc
            As[col * 32 + crow] = (_Float16)Ab[(size_t)(k0 + crow) * HW + m0 + col];
            Bs[col * 32 + crow] = (_Float16)Bg[(size_t)(k0 + crow) * HW + n0 + col];
        }
#endif
        if (k0 + 32 < CC) {  // hint next K tile -> global_prefetch_b8
            __builtin_prefetch(&Ab[(size_t)(k0 + 32) * HW + m0 + (t & 63)], 0, 1);
            __builtin_prefetch(&Bg[(size_t)(k0 + 32) * HW + n0 + (t & 63)], 0, 1);
        }
        __syncthreads();

        union Frag { v16h v; uint4 u[2]; };
        Frag fa[2], fb[2];
        // A fragment (16x32 f16): lanes<16 hold K0-7 & K16-23; lanes>=16 K8-15 & K24-31
#pragma unroll
        for (int mi = 0; mi < 2; ++mi) {
            int row = wm + mi * 16 + l16;
            int kb  = hi * 8;
            fa[mi].u[0] = *(const uint4*)&As[row * 32 + kb];
            fa[mi].u[1] = *(const uint4*)&As[row * 32 + 16 + kb];
        }
        // B fragment (32x16 f16): lanes<16 hold K0-7 & K16-23; lanes>=16 K8-15 & K24-31
#pragma unroll
        for (int ni = 0; ni < 2; ++ni) {
            int col = wn + ni * 16 + l16;
            int kb  = hi * 8;
            fb[ni].u[0] = *(const uint4*)&Bs[col * 32 + kb];
            fb[ni].u[1] = *(const uint4*)&Bs[col * 32 + 16 + kb];
        }
#pragma unroll
        for (int mi = 0; mi < 2; ++mi)
#pragma unroll
            for (int ni = 0; ni < 2; ++ni)
                acc[mi][ni] = __builtin_amdgcn_wmma_f32_16x16x32_f16(
                    false, fa[mi].v, false, fb[ni].v,
                    (short)0, acc[mi][ni], false, false);
        __syncthreads();
    }

    // C layout: VGPR r, lanes 0-15: M=r, N=lane; lanes 16-31: M=r+8, N=lane-16
    float* Cb = M + (size_t)b * HW * HW;
#pragma unroll
    for (int mi = 0; mi < 2; ++mi)
#pragma unroll
        for (int ni = 0; ni < 2; ++ni) {
            int colg = n0 + wn + ni * 16 + l16;
#pragma unroll
            for (int r = 0; r < 8; ++r) {
                int rowg = m0 + wm + mi * 16 + hi * 8 + r;
                Cb[(size_t)rowg * HW + colg] = acc[mi][ni][r];
            }
        }
}

// ---------------------------------------------------------------------------
// Kernel 2: per-row max + sum(exp(alpha*(x-max)))   (softmax axis=-1 stats)
// ---------------------------------------------------------------------------
__global__ __launch_bounds__(256)
void row_stats(const float* __restrict__ M, const float* __restrict__ alphap,
               float* __restrict__ rmax, float* __restrict__ rsum)
{
    __shared__ float sm[256];
    const int rr = blockIdx.x;                  // (b*1600 + i), M contiguous
    const float* row = M + (size_t)rr * HW;
    const float alpha = alphap[0];
    const int t = threadIdx.x;

    float m = -3.4e38f;
    for (int j = t; j < HW; j += 256) m = fmaxf(m, row[j]);
    sm[t] = m; __syncthreads();
    for (int s = 128; s > 0; s >>= 1) { if (t < s) sm[t] = fmaxf(sm[t], sm[t + s]); __syncthreads(); }
    m = sm[0]; __syncthreads();

    float s = 0.f;
    for (int j = t; j < HW; j += 256) s += __expf(alpha * (row[j] - m));
    sm[t] = s; __syncthreads();
    for (int st = 128; st > 0; st >>= 1) { if (t < st) sm[t] += sm[t + st]; __syncthreads(); }
    if (t == 0) { rmax[rr] = m; rsum[rr] = sm[0]; }
}

// ---------------------------------------------------------------------------
// Kernel 3: per-column online-softmax stats (axis=-2). One thread per column;
// consecutive threads touch consecutive columns -> coalesced, L2 resident.
// ---------------------------------------------------------------------------
__global__ __launch_bounds__(256)
void col_stats(const float* __restrict__ M, const float* __restrict__ alphap,
               float* __restrict__ cmax, float* __restrict__ csum)
{
    const int g = blockIdx.x * 256 + threadIdx.x;   // 0..12799
    if (g >= NROW) return;
    const int b = g / HW, j = g % HW;
    const float* base = M + (size_t)b * HW * HW + j;
    const float alpha = alphap[0];
    float m = -3.4e38f, s = 0.f;
    for (int i = 0; i < HW; ++i) {
        float x = base[(size_t)i * HW];
        if (x > m) { s = s * __expf(alpha * (m - x)) + 1.f; m = x; }
        else       { s += __expf(alpha * (x - m)); }
    }
    cmax[g] = m; csum[g] = s;
}

// ---------------------------------------------------------------------------
// Kernel 4: M[i,j] = softmax_row * softmax_col * (1 - gaussian window), in place
// ---------------------------------------------------------------------------
__global__ __launch_bounds__(256)
void make_m(float* __restrict__ M, const float* __restrict__ alphap,
            const float* __restrict__ rmax, const float* __restrict__ rsum,
            const float* __restrict__ cmax, const float* __restrict__ csum)
{
    const size_t total = (size_t)BB * HW * HW;
    const float alpha = alphap[0];
    for (size_t idx = (size_t)blockIdx.x * blockDim.x + threadIdx.x;
         idx < total; idx += (size_t)gridDim.x * blockDim.x) {
        size_t b   = idx / ((size_t)HW * HW);
        size_t rem = idx - b * (size_t)HW * HW;
        int i = (int)(rem / HW), j = (int)(rem % HW);
        float x = M[idx];
        float a = __expf(alpha * (x - rmax[b * HW + i])) / rsum[b * HW + i];
        float c = __expf(alpha * (x - cmax[b * HW + j])) / csum[b * HW + j];
        float dr = (float)(j / 40 - i / 40);
        float dc = (float)(j % 40 - i % 40);
        // sigma = 40*0.15 = 6 -> 2*sigma^2 = 72
        float gau = __expf(-(dr * dr) * (1.f / 72.f)) * __expf(-(dc * dc) * (1.f / 72.f));
        M[idx] = (1.f - gau) * a * c;
    }
}

// ---------------------------------------------------------------------------
// Kernel 5: per-line (row or column of M): Parzen soft-argmax -> grid, and
// iterative top-10 -> val. Line cached in LDS; all passes on-chip.
// ---------------------------------------------------------------------------
__global__ __launch_bounds__(256)
void branch_pass(const float* __restrict__ M, float* __restrict__ grid_out,
                 float* __restrict__ val_out, int colmode)
{
    __shared__ float buf[HW];
    __shared__ float sv[256];
    __shared__ int   si[256];

    const int rr = blockIdx.x;            // 0..12799
    const int b = rr / HW, line = rr % HW;
    const float* base = M + (size_t)b * HW * HW;
    const int t = threadIdx.x;

    if (colmode) { for (int e = t; e < HW; e += 256) buf[e] = base[(size_t)e * HW + line]; }
    else         { for (int e = t; e < HW; e += 256) buf[e] = base[(size_t)line * HW + e]; }
    __syncthreads();

    // hard argmax (first occurrence on ties)
    float bv = -3.4e38f; int bi = 0x7fffffff;
    for (int e = t; e < HW; e += 256) { float v = buf[e]; if (v > bv) { bv = v; bi = e; } }
    sv[t] = bv; si[t] = bi; __syncthreads();
    for (int s = 128; s > 0; s >>= 1) {
        if (t < s) {
            if (sv[t + s] > sv[t] || (sv[t + s] == sv[t] && si[t + s] < si[t])) {
                sv[t] = sv[t + s]; si[t] = si[t + s];
            }
        }
        __syncthreads();
    }
    const float mx = sv[0];
    const int   am = si[0];
    __syncthreads();
    const float cy = (float)(am / 40), cx = (float)(am % 40);

    // Parzen-weighted moments of softmax(5*x)
    float s0 = 0.f, sx = 0.f, sy = 0.f;
    for (int e = t; e < HW; e += 256) {
        float yy = (float)(e / 40), xx = (float)(e % 40);
        float dy = yy - cy, dx = xx - cx;
        float d = sqrtf(dy * dy + dx * dx);
        float r = d * 0.1f;
        float win = 0.f;
        if (d <= 5.f)       win = 1.f - 6.f * r * r * (1.f - r);
        else if (d <= 10.f) { float u = 1.f - r; win = 2.f * u * u * u; }
        float p = __expf(5.f * (buf[e] - mx)) * win;
        s0 += p; sx += p * xx; sy += p * yy;
    }
    sv[t] = s0; __syncthreads();
    for (int s = 128; s > 0; s >>= 1) { if (t < s) sv[t] += sv[t + s]; __syncthreads(); }
    s0 = sv[0]; __syncthreads();
    sv[t] = sx; __syncthreads();
    for (int s = 128; s > 0; s >>= 1) { if (t < s) sv[t] += sv[t + s]; __syncthreads(); }
    sx = sv[0]; __syncthreads();
    sv[t] = sy; __syncthreads();
    for (int s = 128; s > 0; s >>= 1) { if (t < s) sv[t] += sv[t + s]; __syncthreads(); }
    sy = sv[0]; __syncthreads();

    if (t == 0) {
        grid_out[(size_t)rr * 2 + 0] = (sx / s0) * (1.f / 40.f);
        grid_out[(size_t)rr * 2 + 1] = (sy / s0) * (1.f / 40.f);
    }

    // top-10 by iterative max extraction (descending order, matches topk sorted)
    for (int k = 0; k < 10; ++k) {
        float kv = -3.4e38f; int ki = 0;
        for (int e = t; e < HW; e += 256) { float v = buf[e]; if (v > kv) { kv = v; ki = e; } }
        sv[t] = kv; si[t] = ki; __syncthreads();
        for (int s = 128; s > 0; s >>= 1) {
            if (t < s) {
                if (sv[t + s] > sv[t] || (sv[t + s] == sv[t] && si[t + s] < si[t])) {
                    sv[t] = sv[t + s]; si[t] = si[t + s];
                }
            }
            __syncthreads();
        }
        if (t == 0) {
            val_out[(size_t)b * 16000 + (size_t)k * HW + line] = sv[0];
            buf[si[0]] = -3.4e38f;
        }
        __syncthreads();
    }
}

// ---------------------------------------------------------------------------
// Kernel 6: channel concat + bilinear grid_sample (zeros padding, the
// reference's align_corners=False formula with its [0,1]-range grids).
// ---------------------------------------------------------------------------
__global__ __launch_bounds__(256)
void sample_cat(const float* __restrict__ srcCopy,  // copied into channels [0,896)
                const float* __restrict__ srcSamp,  // sampled into channels [896,1792)
                const float* __restrict__ grid,     // (B,1600,2) normalized (x,y)
                float* __restrict__ out)            // base of x?_cat (B,1792,40,40)
{
    const size_t total = (size_t)BB * CC * HW;
    size_t idx = (size_t)blockIdx.x * blockDim.x + threadIdx.x;
    if (idx >= total) return;
    const int p = (int)(idx % HW);
    const size_t bc = idx / HW;
    const int c = (int)(bc % CC);
    const int b = (int)(bc / CC);

    const size_t ob = (size_t)b * 1792 * HW;
    out[ob + (size_t)c * HW + p] = srcCopy[idx];

    const float gx = grid[((size_t)b * HW + p) * 2 + 0];
    const float gy = grid[((size_t)b * HW + p) * 2 + 1];
    const float ix = ((gx + 1.f) * 40.f - 1.f) * 0.5f;
    const float iy = ((gy + 1.f) * 40.f - 1.f) * 0.5f;
    const float ix0 = floorf(ix), iy0 = floorf(iy);
    const float wx1 = ix - ix0, wx0 = 1.f - wx1;
    const float wy1 = iy - iy0, wy0 = 1.f - wy1;
    const int x0 = (int)ix0, y0 = (int)iy0;
    const float* img = srcSamp + (size_t)b * CC * HW + (size_t)c * HW;

    float v = 0.f;
    if (y0     >= 0 && y0     < 40 && x0     >= 0 && x0     < 40) v += wy0 * wx0 * img[y0 * 40 + x0];
    if (y0     >= 0 && y0     < 40 && x0 + 1 >= 0 && x0 + 1 < 40) v += wy0 * wx1 * img[y0 * 40 + x0 + 1];
    if (y0 + 1 >= 0 && y0 + 1 < 40 && x0     >= 0 && x0     < 40) v += wy1 * wx0 * img[(y0 + 1) * 40 + x0];
    if (y0 + 1 >= 0 && y0 + 1 < 40 && x0 + 1 >= 0 && x0 + 1 < 40) v += wy1 * wx1 * img[(y0 + 1) * 40 + x0 + 1];
    out[ob + (size_t)(CC + c) * HW + p] = v;
}

// ---------------------------------------------------------------------------
extern "C" void kernel_launch(void* const* d_in, const int* in_sizes, int n_in,
                              void* d_out, int out_size, void* d_ws, size_t ws_size,
                              hipStream_t stream)
{
    const float* x1    = (const float*)d_in[0];
    const float* x2    = (const float*)d_in[1];
    const float* alpha = (const float*)d_in[2];
    float* out = (float*)d_out;
    float* ws  = (float*)d_ws;

    // workspace layout (floats)
    float* M     = ws;                       // 8*1600*1600 = 20,480,000
    float* rmax  = ws + 20480000;            // 12800
    float* rsum  = rmax + NROW;
    float* cmax  = rsum + NROW;
    float* csum  = cmax + NROW;
    float* grid1 = csum + NROW;              // 2*12800
    float* grid2 = grid1 + 2 * NROW;

    dim3 gg(25, 25, BB);
    corr_gemm_wmma<<<gg, 128, 0, stream>>>(x1, x2, M);
    row_stats<<<NROW, 256, 0, stream>>>(M, alpha, rmax, rsum);
    col_stats<<<NROW / 256, 256, 0, stream>>>(M, alpha, cmax, csum);
    make_m<<<4096, 256, 0, stream>>>(M, alpha, rmax, rsum, cmax, csum);
    branch_pass<<<NROW, 256, 0, stream>>>(M, grid1, out + O_VAL1, 0); // rows  -> grid1, val1
    branch_pass<<<NROW, 256, 0, stream>>>(M, grid2, out + O_VAL2, 1); // cols  -> grid2, val2

    const size_t total = (size_t)BB * CC * HW;
    const int blocks = (int)((total + 255) / 256);
    sample_cat<<<blocks, 256, 0, stream>>>(x1, x2, grid1, out);            // x1_cat
    sample_cat<<<blocks, 256, 0, stream>>>(x2, x1, grid2, out + O_X2CAT);  // x2_cat
}
